// ssm_conv_3040836845601
// MI455X (gfx1250) — matrix operand order repaired and checked by reading, not compiled, not run on hardware
//
#include <hip/hip_runtime.h>
#include <hip/hip_bf16.h>
#include <math.h>

typedef __bf16 bf16_t;
typedef __attribute__((ext_vector_type(16))) __bf16 v16bf;
typedef __attribute__((ext_vector_type(8)))  __bf16 v8bf;
typedef __attribute__((ext_vector_type(8)))  float  v8f;

#define D_MODEL 256
#define D_INNER 512
#define NSTATE  16
#define DTRANK  16
#define LSEQ    4096
#define NBATCH  16
#define MTOK    (NBATCH*LSEQ)   // 65536 tokens

__device__ __forceinline__ float sigmoidf_(float x){ return 1.0f/(1.0f+__expf(-x)); }
__device__ __forceinline__ float siluf_(float x){ return x*sigmoidf_(x); }
__device__ __forceinline__ float softplusf_(float x){ return (x > 20.0f) ? x : log1pf(__expf(x)); }

// CDNA5 async copy: 16 bytes global -> LDS, tracked by ASYNCcnt
__device__ __forceinline__ void async_copy16(uint32_t lds_off, const void* gptr){
    asm volatile("global_load_async_to_lds_b128 %0, %1, off"
                 :: "v"(lds_off), "v"((uint64_t)(uintptr_t)gptr) : "memory");
}
__device__ __forceinline__ void wait_async0(){
    asm volatile("s_wait_asynccnt 0x0" ::: "memory");
}

// ---------------- weight conversion ----------------
__global__ void cvt_f32_bf16(const float* __restrict__ in, bf16_t* __restrict__ out, int n){
    int i = blockIdx.x*blockDim.x + threadIdx.x;
    if (i < n) out[i] = (bf16_t)in[i];
}

// dt_proj_w (512x16) -> bf16 padded to (512x32), cols 16..31 zero
__global__ void pad_dtproj(const float* __restrict__ in, bf16_t* __restrict__ out){
    int i = blockIdx.x*blockDim.x + threadIdx.x;   // 512*32
    int r = i >> 5, c = i & 31;
    out[i] = (c < DTRANK) ? (bf16_t)in[r*DTRANK + c] : (bf16_t)0.0f;
}

// ---------------- LayerNorm (no affine, eps=1e-6), f32 -> bf16 ----------------
__global__ __launch_bounds__(256) void ln_bf16(const float* __restrict__ in, bf16_t* __restrict__ out){
    __shared__ float ssum[8], ssq[8], stat[2];
    int t = threadIdx.x;
    size_t row = blockIdx.x;
    float v = in[row*D_MODEL + t];
    float s = v, q = v*v;
    for (int o = 16; o; o >>= 1){ s += __shfl_xor(s, o, 32); q += __shfl_xor(q, o, 32); }
    if ((t & 31) == 0){ ssum[t>>5] = s; ssq[t>>5] = q; }
    __syncthreads();
    if (t == 0){
        float a = 0.f, b = 0.f;
        for (int i = 0; i < 8; i++){ a += ssum[i]; b += ssq[i]; }
        float mean = a / D_MODEL;
        stat[0] = mean;
        stat[1] = rsqrtf(b / D_MODEL - mean*mean + 1e-6f);
    }
    __syncthreads();
    out[row*D_MODEL + t] = (bf16_t)((v - stat[0]) * stat[1]);
}

// LayerNorm then multiply by bf16 gate -> bf16
__global__ __launch_bounds__(256) void ln_gate_bf16(const float* __restrict__ in,
                                                    const bf16_t* __restrict__ gate,
                                                    bf16_t* __restrict__ out){
    __shared__ float ssum[8], ssq[8], stat[2];
    int t = threadIdx.x;
    size_t row = blockIdx.x;
    float v = in[row*D_MODEL + t];
    float s = v, q = v*v;
    for (int o = 16; o; o >>= 1){ s += __shfl_xor(s, o, 32); q += __shfl_xor(q, o, 32); }
    if ((t & 31) == 0){ ssum[t>>5] = s; ssq[t>>5] = q; }
    __syncthreads();
    if (t == 0){
        float a = 0.f, b = 0.f;
        for (int i = 0; i < 8; i++){ a += ssum[i]; b += ssq[i]; }
        float mean = a / D_MODEL;
        stat[0] = mean;
        stat[1] = rsqrtf(b / D_MODEL - mean*mean + 1e-6f);
    }
    __syncthreads();
    float g = (float)gate[row*D_MODEL + t];
    out[row*D_MODEL + t] = (bf16_t)(((v - stat[0]) * stat[1]) * g);
}

// ---------------- WMMA GEMM: out[M,N] = act(A[M,K] * W[N,K]^T + bias) (+resid) ---------
// Block = 256 threads = 8 waves; block tile = 128 x (NT*16).
// B panel (NT*16 rows x 32 halves per K-step) is staged in LDS with
// global_load_async_to_lds_b128 (double buffered, ASYNCcnt-synchronized); every wave
// then runs NT WMMAs per K-step against its own 16-row A strip.
enum { ACT_NONE = 0, ACT_SIGMOID = 1, ACT_SOFTPLUS = 2 };

template<int ACT, bool OUT_BF16, int NT>
__global__ __launch_bounds__(256)
void gemm_wmma(const bf16_t* __restrict__ A, int lda,
               const bf16_t* __restrict__ W, int ldw,
               const float* __restrict__ bias,
               void* __restrict__ out, int ldo,
               const float* __restrict__ resid,
               int K)
{
    __shared__ bf16_t sB[2][NT*16][32];          // [buf][n][k], 64 B per row
    const int t    = threadIdx.x;
    const int lane = t & 31;
    const int wave = t >> 5;
    const int h = lane >> 4;                     // lane half (ISA 16-bit operand layout)
    const int r = lane & 15;                     // A row in tile / B column in tile
    const int n0 = blockIdx.x * (NT*16);
    const int m0 = (blockIdx.y * 8 + wave) * 16;

    // A 16x32 tile: lane holds row r, contiguous K = h*16..h*16+15
    const bf16_t* arow = A + (size_t)(m0 + r) * lda + h * 16;

    // staging assignment: thread t copies 16 B: row t>>2 of panel, half-chunk (t&3)*8
    const int  crow    = t >> 2;
    const int  ccol    = (t & 3) * 8;
    const bool do_copy = crow < NT*16;
    const bf16_t* wrow = W + (size_t)(n0 + crow) * ldw + ccol;

    v8f acc[NT];
    #pragma unroll
    for (int i = 0; i < NT; i++){ v8f z = {}; acc[i] = z; }

    // prologue: stage K-chunk 0 into buffer 0
    if (do_copy)
        async_copy16((uint32_t)(uintptr_t)&sB[0][crow][ccol], wrow);
    wait_async0();
    __syncthreads();

    int buf = 0;
    for (int k0 = 0; k0 < K; k0 += 32){
        if (k0 + 32 < K){
            if (do_copy)
                async_copy16((uint32_t)(uintptr_t)&sB[buf ^ 1][crow][ccol], wrow + k0 + 32);
            __builtin_prefetch(arow + k0 + 64, 0, 1);     // A chunk two K-steps ahead (gfx1250 prefetch)
        }
        v8bf alo = *(const v8bf*)(arow + k0);
        v8bf ahi = *(const v8bf*)(arow + k0 + 8);
        v16bf am = __builtin_shufflevector(alo, ahi,
                      0,1,2,3,4,5,6,7,8,9,10,11,12,13,14,15);
        #pragma unroll
        for (int nt = 0; nt < NT; nt++){
            // B 32x16 tile: lane holds col r, contiguous K = h*16..h*16+15 (from LDS)
            v16bf bm = *(const v16bf*)&sB[buf][nt*16 + r][h*16];
            acc[nt] = __builtin_amdgcn_wmma_f32_16x16x32_bf16(false, am, false, bm,
                                                              (short)0, acc[nt], false, false);
        }
        wait_async0();
        __syncthreads();
        buf ^= 1;
    }

    #pragma unroll
    for (int nt = 0; nt < NT; nt++){
        const int n = n0 + nt*16 + r;
        const float bv = bias ? bias[n] : 0.0f;
        #pragma unroll
        for (int j = 0; j < 8; j++){
            int m = m0 + h * 8 + j;              // C/D layout: VGPR j -> row j (+8 hi lanes)
            float v = acc[nt][j] + bv;
            if (ACT == ACT_SIGMOID)       v = sigmoidf_(v);
            else if (ACT == ACT_SOFTPLUS) v = softplusf_(v);
            size_t idx = (size_t)m * ldo + n;
            if (resid) v += resid[idx];
            if (OUT_BF16) ((bf16_t*)out)[idx] = (bf16_t)v;
            else          ((float*)out)[idx]  = v;
        }
    }
}

// ---------------- causal depthwise conv1d (d_conv=4) + bias + SiLU -------------
__global__ __launch_bounds__(256) void conv_silu(const bf16_t* __restrict__ xz,
                                                 const float* __restrict__ cw,
                                                 const float* __restrict__ cb,
                                                 bf16_t* __restrict__ xact){
    size_t i = (size_t)blockIdx.x * blockDim.x + threadIdx.x;   // MTOK*512
    int c = (int)(i & (D_INNER - 1));
    size_t m = i >> 9;
    int l = (int)(m & (LSEQ - 1));
    const bf16_t* base = xz + m * (2*D_INNER) + c;
    float acc = cb[c];
    #pragma unroll
    for (int j = 0; j < 4; j++){
        int dl = l - 3 + j;
        if (dl >= 0)
            acc += cw[c*4 + j] * (float)base[(ptrdiff_t)(j - 3) * (2*D_INNER)];
    }
    xact[i] = (bf16_t)siluf_(acc);
}

// dt (cols 0..15 of x_dbl f32, row stride 48) -> bf16 (M x 32), zero padded
__global__ __launch_bounds__(256) void prep_dt(const float* __restrict__ xdbl, bf16_t* __restrict__ dtp){
    size_t i = (size_t)blockIdx.x * blockDim.x + threadIdx.x;   // MTOK*32
    size_t m = i >> 5; int c = (int)(i & 31);
    dtp[i] = (c < DTRANK) ? (bf16_t)xdbl[m*48 + c] : (bf16_t)0.0f;
}

// ---------------- selective scan + skip (x*D) + SiLU(z) gate -------------------
__global__ __launch_bounds__(512)
void ssm_scan(const bf16_t* __restrict__ delta, const bf16_t* __restrict__ xact,
              const float* __restrict__ xdbl,  const bf16_t* __restrict__ xz,
              const float* __restrict__ A_log, const float* __restrict__ Dp,
              bf16_t* __restrict__ yg)
{
    __shared__ float sB[NSTATE], sC[NSTATE];
    const int c = threadIdx.x;
    const int b = blockIdx.x;
    float a[NSTATE], hst[NSTATE];
    #pragma unroll
    for (int n = 0; n < NSTATE; n++){ a[n] = -__expf(A_log[c*NSTATE + n]); hst[n] = 0.0f; }
    const float Dc = Dp[c];

    for (int l = 0; l < LSEQ; l++){
        size_t m = (size_t)b * LSEQ + l;
        if (c < 2*NSTATE){
            float v = xdbl[m*48 + DTRANK + c];
            if (c < NSTATE) sB[c] = v; else sC[c - NSTATE] = v;
        }
        __syncthreads();
        float dt = (float)delta[m*D_INNER + c];
        float xv = (float)xact[m*D_INNER + c];
        float dx = dt * xv;
        float y = 0.0f;
        #pragma unroll
        for (int n = 0; n < NSTATE; n++){
            float dA = __expf(dt * a[n]);
            hst[n] = dA * hst[n] + dx * sB[n];
            y += hst[n] * sC[n];
        }
        float zc = (float)xz[m*(2*D_INNER) + D_INNER + c];
        y = (y + xv * Dc) * siluf_(zc);
        yg[m*D_INNER + c] = (bf16_t)y;
        __syncthreads();
    }
}

// ---------------- launcher -----------------------------------------------------
extern "C" void kernel_launch(void* const* d_in, const int* in_sizes, int n_in,
                              void* d_out, int out_size, void* d_ws, size_t ws_size,
                              hipStream_t stream)
{
    (void)in_sizes; (void)n_in; (void)out_size; (void)ws_size;
    const float* x        = (const float*)d_in[0];
    const float* w1       = (const float*)d_in[1];
    const float* b1       = (const float*)d_in[2];
    const float* w2       = (const float*)d_in[3];
    const float* b2       = (const float*)d_in[4];
    const float* w3       = (const float*)d_in[5];
    const float* b3       = (const float*)d_in[6];
    const float* in_projw = (const float*)d_in[7];
    const float* conv_w   = (const float*)d_in[8];
    const float* conv_b   = (const float*)d_in[9];
    const float* x_projw  = (const float*)d_in[10];
    const float* dt_projw = (const float*)d_in[11];
    const float* dt_projb = (const float*)d_in[12];
    const float* A_log    = (const float*)d_in[13];
    const float* Dvec     = (const float*)d_in[14];
    const float* out_projw= (const float*)d_in[15];

    char* ws = (char*)d_ws;
    size_t off = 0;
    auto alloc = [&](size_t bytes) -> void* {
        void* p = ws + off;
        off += (bytes + 255) & ~(size_t)255;
        return p;
    };

    bf16_t* wb1   = (bf16_t*)alloc((size_t)256*256*2);
    bf16_t* wb2   = (bf16_t*)alloc((size_t)256*256*2);
    bf16_t* wb3   = (bf16_t*)alloc((size_t)256*256*2);
    bf16_t* wbin  = (bf16_t*)alloc((size_t)1024*256*2);
    bf16_t* wbx   = (bf16_t*)alloc((size_t)48*512*2);
    bf16_t* wbdt  = (bf16_t*)alloc((size_t)512*32*2);
    bf16_t* wbout = (bf16_t*)alloc((size_t)256*512*2);
    bf16_t* xn    = (bf16_t*)alloc((size_t)MTOK*256*2);
    bf16_t* xx1   = (bf16_t*)alloc((size_t)MTOK*256*2);
    bf16_t* xx2   = (bf16_t*)alloc((size_t)MTOK*256*2);
    bf16_t* xzb   = (bf16_t*)alloc((size_t)MTOK*1024*2);
    bf16_t* xact  = (bf16_t*)alloc((size_t)MTOK*512*2);
    float*  xdbl  = (float*) alloc((size_t)MTOK*48*4);
    bf16_t* dtp   = (bf16_t*)alloc((size_t)MTOK*32*2);
    bf16_t* dlt   = (bf16_t*)alloc((size_t)MTOK*512*2);
    bf16_t* yg    = (bf16_t*)alloc((size_t)MTOK*512*2);
    float*  mo    = (float*) alloc((size_t)MTOK*256*4);
    bf16_t* a2    = (bf16_t*)alloc((size_t)MTOK*256*2);

    // weights -> bf16
    cvt_f32_bf16<<<(256*256+255)/256,  256, 0, stream>>>(w1, wb1, 256*256);
    cvt_f32_bf16<<<(256*256+255)/256,  256, 0, stream>>>(w2, wb2, 256*256);
    cvt_f32_bf16<<<(256*256+255)/256,  256, 0, stream>>>(w3, wb3, 256*256);
    cvt_f32_bf16<<<(1024*256+255)/256, 256, 0, stream>>>(in_projw, wbin, 1024*256);
    cvt_f32_bf16<<<(48*512+255)/256,   256, 0, stream>>>(x_projw, wbx, 48*512);
    cvt_f32_bf16<<<(256*512+255)/256,  256, 0, stream>>>(out_projw, wbout, 256*512);
    pad_dtproj<<<(512*32)/256, 256, 0, stream>>>(dt_projw, wbdt);

    // LN(x) -> bf16
    ln_bf16<<<MTOK, 256, 0, stream>>>(x, xn);

    dim3 blk(256);
    dim3 gNM(D_MODEL/64, MTOK/128);                  // N=256, NT=4
    gemm_wmma<ACT_SIGMOID, true, 4><<<gNM, blk, 0, stream>>>(xn, 256, wb1, 256, b1, xx1, 256, nullptr, 256);
    gemm_wmma<ACT_SIGMOID, true, 4><<<gNM, blk, 0, stream>>>(xn, 256, wb2, 256, b2, xx2, 256, nullptr, 256);

    // xz = xx1 @ in_proj^T  (M x 1024)
    dim3 gIN(1024/64, MTOK/128);
    gemm_wmma<ACT_NONE, true, 4><<<gIN, blk, 0, stream>>>(xx1, 256, wbin, 256, nullptr, xzb, 1024, nullptr, 256);

    // causal depthwise conv + SiLU on xm half
    conv_silu<<<((size_t)MTOK*512)/256, 256, 0, stream>>>(xzb, conv_w, conv_b, xact);

    // x_dbl = xact @ x_proj^T  (M x 48, f32)  -- NT=3 covers N=48 in one block column
    dim3 gXP(1, MTOK/128);
    gemm_wmma<ACT_NONE, false, 3><<<gXP, blk, 0, stream>>>(xact, 512, wbx, 512, nullptr, xdbl, 48, nullptr, 512);

    // dt padded to K=32, then delta = softplus(dt @ dt_proj^T + b)  (M x 512)
    prep_dt<<<((size_t)MTOK*32)/256, 256, 0, stream>>>(xdbl, dtp);
    dim3 gDT(512/64, MTOK/128);
    gemm_wmma<ACT_SOFTPLUS, true, 4><<<gDT, blk, 0, stream>>>(dtp, 32, wbdt, 32, dt_projb, dlt, 512, nullptr, 32);

    // selective scan + D skip + SiLU(z) gate
    ssm_scan<<<NBATCH, 512, 0, stream>>>(dlt, xact, xdbl, xzb, A_log, Dvec, yg);

    // m_out = yg @ out_proj^T  (M x 256, f32)
    gemm_wmma<ACT_NONE, false, 4><<<gNM, blk, 0, stream>>>(yg, 512, wbout, 512, nullptr, mo, 256, nullptr, 512);

    // a2 = LN(m_out) * xx2 -> bf16
    ln_gate_bf16<<<MTOK, 256, 0, stream>>>(mo, xx2, a2);

    // out = a2 @ w3^T + b3 + x  (f32)
    gemm_wmma<ACT_NONE, false, 4><<<gNM, blk, 0, stream>>>(a2, 256, wb3, 256, b3, (float*)d_out, 256, x, 256);
}
